// EdgeModel_7713761264051
// MI455X (gfx1250) — matrix operand order repaired and checked by reading, not compiled, not run on hardware
//
#include <hip/hip_runtime.h>
#include <hip/hip_bf16.h>

typedef __attribute__((ext_vector_type(16))) __bf16 v16bf;
typedef __attribute__((ext_vector_type(8)))  __bf16 v8bf;
typedef __attribute__((ext_vector_type(16))) float  v16f;
typedef __attribute__((ext_vector_type(8)))  float  v8f;

union BF16x16 { __bf16 h[16]; unsigned u[8]; v16bf v; };
union BF16x8  { v8bf v; uint4 q; };
union F32x16  { float f[16]; v16f v; };
union F32x8   { float f[8];  v8f  v; };

// RNE f32 -> bf16 pair packed into one dword (weight staging path).
__device__ __forceinline__ unsigned bf2pk(float lo, float hi) {
    typedef __attribute__((ext_vector_type(2))) float  v2f;
    typedef __attribute__((ext_vector_type(2))) __bf16 v2bf;
    union { v2bf v; unsigned u; } r;
    v2f t; t.x = lo; t.y = hi;
    r.v = __builtin_convertvector(t, v2bf);
    return r.u;
}

// Build a 16x32 bf16 A-fragment for this lane from fp32 memory.
// p points at row_base + k_tile_base + 8*hi; lane holds K runs [0..7] -> elems
// 0..7 and [16..23] -> elems 8..15 (per the ISA 16-bit A layout).
__device__ __forceinline__ v16bf load_a_frag(const float* __restrict__ p) {
    const float4* q = (const float4*)p;
    F32x16 t;
    *(float4*)&t.f[0]  = q[0];
    *(float4*)&t.f[4]  = q[1];
    *(float4*)&t.f[8]  = q[4];
    *(float4*)&t.f[12] = q[5];
    return __builtin_convertvector(t.v, v16bf);
}

// Read a pre-swizzled B fragment (32 bytes per lane) from LDS.
__device__ __forceinline__ v16bf lds_frag(const unsigned* p) {
    BF16x16 r;
#pragma unroll
    for (int i = 0; i < 8; ++i) r.u[i] = p[i];
    return r.v;
}

#define NWAVES 8   // 256 threads / wave32
#define MT     2   // M-tiles (16 edges each) per wave iteration -> B reuse x2

__global__ __launch_bounds__(256)
void edge_mlp_kernel(const float* __restrict__ src,
                     const float* __restrict__ dst,
                     const float* __restrict__ ea,
                     const float* __restrict__ w1,   // [160,128] row-major
                     const float* __restrict__ b1,   // [128]
                     const float* __restrict__ w2,   // [128,32] row-major
                     const float* __restrict__ b2,   // [32]
                     float* __restrict__ out,        // [E,32]
                     int E, int NT)                  // NT = 32-edge tiles
{
    // W1: 5 K-tiles x 8 N-tiles = 40 fragments, 1KB each (32 lanes x 32B)
    __shared__ unsigned w1f[40 * 32 * 8];            // 40 KB
    // W2: 4 K-tiles x 2 N-tiles = 8 fragments
    __shared__ unsigned w2f[8 * 32 * 8];             // 8 KB
    // h staging: per wave, MT tiles of h^T [k(0..127)][m(0..15)] bf16 = 4KB ea.
    __shared__ unsigned hst[NWAVES * MT * 128 * 8];  // 64 KB
    __shared__ float    b1s[128];

    const int tid = threadIdx.x;

    // ---- Stage W1 into B-fragment layout (B: lane = N + 16*((K>>3)&1), elem = (K&7) + 8*(K>>4))
    for (int d = tid; d < 40 * 32 * 8; d += 256) {
        int chunk = d >> 3, i = d & 7;          // i = dword within lane's 8
        int lane = chunk & 31, fidx = chunk >> 5;
        int fj = fidx & 7, ft = fidx >> 3;      // n-tile, k-tile
        int hiS = (lane >> 4) & 1, hlS = lane & 15;
        int n  = fj * 16 + hlS;
        int k0 = ft * 32 + hiS * 8 + 2 * i + (i >> 2) * 8;    // elems 2i, 2i+1 -> K k0, k0+1
        w1f[d] = bf2pk(w1[k0 * 128 + n], w1[(k0 + 1) * 128 + n]);
    }
    // ---- Stage W2 likewise
    for (int d = tid; d < 8 * 32 * 8; d += 256) {
        int chunk = d >> 3, i = d & 7;
        int lane = chunk & 31, fidx = chunk >> 5;
        int fj = fidx & 1, fc = fidx >> 1;
        int hiS = (lane >> 4) & 1, hlS = lane & 15;
        int n  = fj * 16 + hlS;
        int k0 = fc * 32 + hiS * 8 + 2 * i + (i >> 2) * 8;
        w2f[d] = bf2pk(w2[k0 * 32 + n], w2[(k0 + 1) * 32 + n]);
    }
    if (tid < 128) b1s[tid] = b1[tid];
    __syncthreads();

    const int lane = tid & 31;
    const int hl   = lane & 15;
    const int hi   = lane >> 4;
    const int wib  = tid >> 5;
    unsigned* hw  = hst + wib * (MT * 128 * 8);      // per-wave 8KB region
    const __bf16* hwb = (const __bf16*)hw;

    // b2 bias per lane (C/D fragment: n = hl for every VGPR)
    const float b2v0 = b2[hl];
    const float b2v1 = b2[16 + hl];

    const int totalWaves = (int)gridDim.x * NWAVES;
    const int wgid = (int)blockIdx.x * NWAVES + wib;

    for (int tile = wgid; tile < NT; tile += totalWaves) {
        const int e0 = tile * 32;                    // 32 edges per iteration

        // ---- A fragments for both 16-edge M-tiles
        v16bf A[MT][5];
#pragma unroll
        for (int u = 0; u < MT; ++u) {
            int row = e0 + u * 16 + hl;
            if (row >= E) row = E - 1;               // clamp (tail only)
            const float* sr = src + row * 64;
            const float* dr = dst + row * 64;
            const float* er = ea  + row * 32;
            A[u][0] = load_a_frag(sr +  0 + 8 * hi);
            A[u][1] = load_a_frag(sr + 32 + 8 * hi);
            A[u][2] = load_a_frag(dr +  0 + 8 * hi);
            A[u][3] = load_a_frag(dr + 32 + 8 * hi);
            A[u][4] = load_a_frag(er +  0 + 8 * hi);
        }

        // Prefetch next tile's rows (global_prefetch_b8)
        int ntile = tile + totalWaves;
        if (ntile < NT) {
            int nr = ntile * 32 + hl;
            __builtin_prefetch(src + nr * 64, 0, 0);
            __builtin_prefetch(dst + nr * 64, 0, 0);
            __builtin_prefetch(ea  + nr * 32, 0, 0);
        }

        // ---- GEMM1: h = relu(x*W1 + b1); each B fragment reused for 2 WMMAs.
        // NOT unrolled: keeps B-fragment live range short -> no spills.
#pragma unroll 1
        for (int j = 0; j < 8; ++j) {
            float bias = b1s[j * 16 + hl];
            F32x8 acc[MT];
#pragma unroll
            for (int u = 0; u < MT; ++u)
#pragma unroll
                for (int i = 0; i < 8; ++i) acc[u].f[i] = bias;
#pragma unroll
            for (int t = 0; t < 5; ++t) {
                v16bf Bf = lds_frag(&w1f[(t * 8 + j) * 256 + lane * 8]);
#pragma unroll
                for (int u = 0; u < MT; ++u)
                    acc[u].v = __builtin_amdgcn_wmma_f32_16x16x32_bf16(
                        false, A[u][t], false, Bf, (short)0, acc[u].v,
                        false, false);
            }
#pragma unroll
            for (int u = 0; u < MT; ++u) {
#pragma unroll
                for (int i = 0; i < 8; ++i)
                    acc[u].f[i] = fmaxf(acc[u].f[i], 0.0f);
                // C-frag lane holds m = 8*hi + r at fixed n = 16j+hl.
                // Store h^T[n][m]: 8 contiguous bf16 -> one ds_store_b128.
                BF16x8 pk;
                pk.v = __builtin_convertvector(acc[u].v, v8bf);
                *(uint4*)(hw + u * 1024 + (j * 16 + hl) * 8 + hi * 4) = pk.q;
            }
        }

        // ---- Re-read h as A-fragments for GEMM2 (K = 128, 4 K-tiles)
        v16bf A2[MT][4];
#pragma unroll
        for (int u = 0; u < MT; ++u) {
            const __bf16* hb = hwb + u * 2048;
#pragma unroll
            for (int c = 0; c < 4; ++c) {
                BF16x16 r;
#pragma unroll
                for (int d = 0; d < 8; ++d) {
                    int K = c * 32 + ((d < 4) ? (8 * hi + 2 * d)
                                              : (16 + 8 * hi + 2 * (d - 4)));
                    r.h[2 * d]     = hb[K * 16 + hl];
                    r.h[2 * d + 1] = hb[(K + 1) * 16 + hl];
                }
                A2[u][c] = r.v;
            }
        }

        // ---- GEMM2: out = h*W2 + b2; B fragment shared across both M-tiles.
#pragma unroll
        for (int j = 0; j < 2; ++j) {
            float bias = (j == 0) ? b2v0 : b2v1;
            F32x8 acc[MT];
#pragma unroll
            for (int u = 0; u < MT; ++u)
#pragma unroll
                for (int i = 0; i < 8; ++i) acc[u].f[i] = bias;
#pragma unroll
            for (int c = 0; c < 4; ++c) {
                v16bf Bf = lds_frag(&w2f[(c * 2 + j) * 256 + lane * 8]);
#pragma unroll
                for (int u = 0; u < MT; ++u)
                    acc[u].v = __builtin_amdgcn_wmma_f32_16x16x32_bf16(
                        false, A2[u][c], false, Bf, (short)0, acc[u].v,
                        false, false);
            }
#pragma unroll
            for (int u = 0; u < MT; ++u)
#pragma unroll
                for (int r = 0; r < 8; ++r) {
                    int m = e0 + u * 16 + 8 * hi + r;
                    if (m < E) out[m * 32 + j * 16 + hl] = acc[u].f[r];
                }
        }
    }
}

extern "C" void kernel_launch(void* const* d_in, const int* in_sizes, int n_in,
                              void* d_out, int out_size, void* d_ws, size_t ws_size,
                              hipStream_t stream) {
    // setup_inputs order: src, dest, edge_attr, u, batch, w1, b1, w2, b2
    const float* src = (const float*)d_in[0];
    const float* dst = (const float*)d_in[1];
    const float* ea  = (const float*)d_in[2];
    // d_in[3] (u) and d_in[4] (batch) are unused by the reference math.
    const float* w1  = (const float*)d_in[5];
    const float* b1  = (const float*)d_in[6];
    const float* w2  = (const float*)d_in[7];
    const float* b2  = (const float*)d_in[8];
    float* out = (float*)d_out;

    int E  = in_sizes[0] / 64;          // src is [E, 64]
    int NT = (E + 31) / 32;             // 32-edge tiles (E=1.6M -> 50000)

    int blocks = (NT + NWAVES - 1) / NWAVES;
    if (blocks > 1250) blocks = 1250;   // grid-stride: 5 iters/wave at E=1.6M

    edge_mlp_kernel<<<blocks, 256, 0, stream>>>(src, dst, ea, w1, b1, w2, b2,
                                                out, E, NT);
}
